// DHGCNLayer_69063074119705
// MI455X (gfx1250) — matrix operand order repaired and checked by reading, not compiled, not run on hardware
//
#include <hip/hip_runtime.h>
#include <hip/hip_bf16.h>

// ---------------------------------------------------------------------------
// DHGCN layer for MI455X (gfx1250, wave32, WMMA + Tensor Data Mover).
// All GEMM operands staged into LDS by TENSOR_LOAD_TO_LDS (TDM, TENSORcnt),
// with hardware LDS padding reproducing the bank-padded layout; compute via
// v_wmma_f32_16x16x32_f16. B operands are materialized transposed ([N][K])
// so every tile is a plain 2D strided TDM descriptor.
// k-means is incremental per-cluster distance sums, one persistent workgroup.
// ---------------------------------------------------------------------------

typedef _Float16 v16h __attribute__((ext_vector_type(16)));
typedef _Float16 v8h  __attribute__((ext_vector_type(8)));
typedef float    v8f  __attribute__((ext_vector_type(8)));
typedef unsigned uv4  __attribute__((ext_vector_type(4)));
typedef unsigned uv8  __attribute__((ext_vector_type(8)));

#define TILE_M 64
#define TILE_N 128
#define TILE_K 32
#define LDSP   40   // padded LDS row stride in halves (80B): 16 DW row + 4 DW pad

// ---------------------------------------------------------------------------
// TDM 2D tile load: global (row-major, strideElems per row) -> LDS with
// hardware padding (pad_interval = 16 DWORDs, pad_amount = 4 DWORDs, i.e.
// 32-half rows land at stride-40-half rows). Rows >= dim1Rows are zero-filled
// by the TDM OOB rule. D# layout per CDNA5 ISA 8.3/8.4 (2D: groups 0-1 only).
// ---------------------------------------------------------------------------
__device__ __forceinline__ void tdm_load_2d(void* ldsPtr, const void* gPtr,
                                            unsigned dim1Rows, unsigned tileRows,
                                            unsigned strideElems) {
  unsigned ldsOff = (unsigned)(unsigned long long)(uintptr_t)ldsPtr;
  unsigned long long ga = (unsigned long long)(uintptr_t)gPtr;
  uv4 g0;
  g0[0] = 1u;                                    // count=1 (valid user D#)
  g0[1] = ldsOff;                                // lds_addr (bytes)
  g0[2] = (unsigned)ga;                          // global_addr[31:0]
  g0[3] = (unsigned)(ga >> 32) | (2u << 30);     // global_addr[56:32] | type=2
  uv8 g1;
  g1[0] = (1u << 16)                             // data_size = 2 bytes
        | (1u << 20)                             // pad_enable
        | (3u << 22)                             // pad_interval: 16 DWORDs
        | (3u << 25);                            // pad_amount: 4 DWORDs
  g1[1] = 32u << 16;                             // tensor_dim0 = 32 (lo16)
  g1[2] = (dim1Rows & 0xFFFFu) << 16;            // tensor_dim0 hi=0 | dim1 lo16
  g1[3] = (dim1Rows >> 16) | (32u << 16);        // dim1 hi16 | tile_dim0 = 32
  g1[4] = tileRows;                              // tile_dim1 | tile_dim2=0
  g1[5] = strideElems;                           // tensor_dim0_stride[31:0]
  g1[6] = 0u;                                    // stride hi | dim1_stride lo
  g1[7] = 0u;
  asm volatile("tensor_load_to_lds %0, %1" :: "s"(g0), "s"(g1) : "memory");
}

// ---------------------------------------------------------------------------
// f32 -> f16 convert (row-major copy)
// ---------------------------------------------------------------------------
__global__ void cvt_f32_f16(const float* __restrict__ src, _Float16* __restrict__ dst, int n) {
  int i = blockIdx.x * 256 + threadIdx.x;
  if (i < n) dst[i] = (_Float16)src[i];
}

// ---------------------------------------------------------------------------
// fused convert + transpose: dst[N][K] f16 from src[K][N] f32
// ---------------------------------------------------------------------------
__global__ void cvt_transpose_f32_f16(const float* __restrict__ src,
                                      _Float16* __restrict__ dst, int rows, int cols) {
  __shared__ float t[16][17];
  int j = blockIdx.x * 16 + threadIdx.x;   // col
  int i = blockIdx.y * 16 + threadIdx.y;   // row
  if (i < rows && j < cols) t[threadIdx.y][threadIdx.x] = src[(size_t)i * cols + j];
  __syncthreads();
  int oi = blockIdx.x * 16 + threadIdx.y;  // col -> dst row
  int oj = blockIdx.y * 16 + threadIdx.x;  // row -> dst col
  if (oi < cols && oj < rows)
    dst[(size_t)oi * rows + oj] = (_Float16)t[threadIdx.x][threadIdx.y];
}

// ---------------------------------------------------------------------------
// f16 transpose (tiled): dst[c*rows + r] = src[r*cols + c]
// ---------------------------------------------------------------------------
__global__ void transpose_h(const _Float16* __restrict__ src, _Float16* __restrict__ dst,
                            int rows, int cols) {
  __shared__ _Float16 t[16][17];
  int j = blockIdx.x * 16 + threadIdx.x;
  int i = blockIdx.y * 16 + threadIdx.y;
  if (i < rows && j < cols) t[threadIdx.y][threadIdx.x] = src[(size_t)i * cols + j];
  __syncthreads();
  int oi = blockIdx.x * 16 + threadIdx.y;
  int oj = blockIdx.y * 16 + threadIdx.x;
  if (oi < cols && oj < rows) dst[(size_t)oi * rows + oj] = t[threadIdx.x][threadIdx.y];
}

// ---------------------------------------------------------------------------
// zero-fill f16
// ---------------------------------------------------------------------------
__global__ void zero_f16(_Float16* __restrict__ dst, int n) {
  int i = blockIdx.x * 256 + threadIdx.x;
  if (i < n) dst[i] = (_Float16)0.f;
}

// ---------------------------------------------------------------------------
// fragment helpers (CDNA5 16-bit WMMA VGPR layouts, ISA 7.12.2)
// A 16x32: halves 0-7 = K[kg*8 .. +7], halves 8-15 = K[kg*8+16 .. +23]
// B 32x16 (LDS tile is [n][k]): halves 0-7 = K[kg*8 .. +7], halves 8-15 = K[kg*8+16 .. +23]
// ---------------------------------------------------------------------------
__device__ __forceinline__ v16h ldsAfrag(const _Float16* rowp, int kg) {
  v8h lo = *(const v8h*)(rowp + kg * 8);
  v8h hi = *(const v8h*)(rowp + kg * 8 + 16);
  return __builtin_shufflevector(lo, hi, 0, 1, 2, 3, 4, 5, 6, 7,
                                 8, 9, 10, 11, 12, 13, 14, 15);
}
__device__ __forceinline__ v16h ldsBfrag(const _Float16* colp, int kg) {
  v8h lo = *(const v8h*)(colp + kg * 8);
  v8h hi = *(const v8h*)(colp + kg * 8 + 16);
  return __builtin_shufflevector(lo, hi, 0, 1, 2, 3, 4, 5, 6, 7,
                                 8, 9, 10, 11, 12, 13, 14, 15);
}
__device__ __forceinline__ v8f wmma16(v16h a, v16h b, v8f c) {
  return __builtin_amdgcn_wmma_f32_16x16x32_f16(false, a, false, b, (short)0, c,
                                                false, false);
}

// ---------------------------------------------------------------------------
// WMMA GEMM: C[M,N] = A[M,K] @ B[K,N], with B supplied TRANSPOSED as
// BT[N][K] row-major. Both tiles staged by TDM (wave 0 issues, TENSORcnt),
// double-buffered LDS, one barrier per K-step, 4 WMMAs per wave per step.
// Requires: K % 32 == 0, N % 128 == 0 (holds at all call sites).
// ---------------------------------------------------------------------------
template <bool OUT_F16, bool ADD_BIAS>
__launch_bounds__(256)
__global__ void gemm_wmma_f16(const _Float16* __restrict__ A,
                              const _Float16* __restrict__ BT,
                              const float* __restrict__ bias,
                              void* __restrict__ Cout,
                              int M, int N, int K) {
  __shared__ _Float16 As[2][TILE_M][LDSP];
  __shared__ _Float16 Bs[2][TILE_N][LDSP];   // [n][k]

  const int tid   = threadIdx.x;
  const int lane  = tid & 31;
  const int wid   = tid >> 5;   // 0..7
  const int waveM = wid >> 2;   // 0..1 -> 32 rows
  const int waveN = wid & 3;    // 0..3 -> 32 cols
  const int kg    = lane >> 4;  // 0/1
  const int mr    = lane & 15;

  const int mBase = blockIdx.y * TILE_M;
  const int nBase = blockIdx.x * TILE_N;

  v8f acc00 = {}, acc01 = {}, acc10 = {}, acc11 = {};

  const unsigned aRows = (unsigned)(M - mBase);  // TDM zero-fills rows >= aRows
  const unsigned bRows = (unsigned)(N - nBase);
  const _Float16* aTile = A + (size_t)mBase * K;
  const _Float16* bTile = BT + (size_t)nBase * K;
  const int nt = K / TILE_K;

  // -------- prologue: TDM-stage tile 0 into buffer 0 --------
  if (wid == 0) {
    tdm_load_2d(&As[0][0][0], aTile, aRows, TILE_M, (unsigned)K);
    tdm_load_2d(&Bs[0][0][0], bTile, bRows, TILE_N, (unsigned)K);
    __builtin_amdgcn_s_wait_tensorcnt(0);
  }
  __syncthreads();

  for (int kt = 0; kt < nt; ++kt) {
    const int p = kt & 1;
    const bool more = (kt + 1) < nt;

    if (more && wid == 0) {
      // overlap next-tile TDM with this tile's WMMAs
      tdm_load_2d(&As[p ^ 1][0][0], aTile + (size_t)(kt + 1) * TILE_K,
                  aRows, TILE_M, (unsigned)K);
      tdm_load_2d(&Bs[p ^ 1][0][0], bTile + (size_t)(kt + 1) * TILE_K,
                  bRows, TILE_N, (unsigned)K);
    }

    v16h a0 = ldsAfrag(&As[p][waveM * 32 + mr][0], kg);
    v16h a1 = ldsAfrag(&As[p][waveM * 32 + 16 + mr][0], kg);
    v16h b0 = ldsBfrag(&Bs[p][waveN * 32 + mr][0], kg);
    v16h b1 = ldsBfrag(&Bs[p][waveN * 32 + 16 + mr][0], kg);

    acc00 = wmma16(a0, b0, acc00);
    acc01 = wmma16(a0, b1, acc01);
    acc10 = wmma16(a1, b0, acc10);
    acc11 = wmma16(a1, b1, acc11);

    if (more && wid == 0) __builtin_amdgcn_s_wait_tensorcnt(0);
    __syncthreads();
  }

  // store: element v of v8f -> row (v + 8*(lane>>4)), col (lane&15)
  const int col0 = nBase + waveN * 32 + mr;
  const int col1 = col0 + 16;
  float bv0 = 0.f, bv1 = 0.f;
  if constexpr (ADD_BIAS) { bv0 = bias[col0]; bv1 = bias[col1]; }
#pragma unroll
  for (int v = 0; v < 8; ++v) {
    int r0 = mBase + waveM * 32 + (lane >> 4) * 8 + v;
    int r1 = r0 + 16;
    if (r0 < M) {
      float x0 = acc00[v] + bv0, x1 = acc01[v] + bv1;
      if constexpr (OUT_F16) {
        ((_Float16*)Cout)[(size_t)r0 * N + col0] = (_Float16)x0;
        ((_Float16*)Cout)[(size_t)r0 * N + col1] = (_Float16)x1;
      } else {
        ((float*)Cout)[(size_t)r0 * N + col0] = x0;
        ((float*)Cout)[(size_t)r0 * N + col1] = x1;
      }
    }
    if (r1 < M) {
      float x0 = acc10[v] + bv0, x1 = acc11[v] + bv1;
      if constexpr (OUT_F16) {
        ((_Float16*)Cout)[(size_t)r1 * N + col0] = (_Float16)x0;
        ((_Float16*)Cout)[(size_t)r1 * N + col1] = (_Float16)x1;
      } else {
        ((float*)Cout)[(size_t)r1 * N + col0] = x0;
        ((float*)Cout)[(size_t)r1 * N + col1] = x1;
      }
    }
  }
}

// ---------------------------------------------------------------------------
// Distances from Gram matrix: D[i,j] = sqrt(max(G[i,i] + G[j,j] - 2 G[i,j], 0))
// ---------------------------------------------------------------------------
__global__ void dist_from_gram(const float* __restrict__ G, float* __restrict__ D, int n) {
  int idx = blockIdx.x * 256 + threadIdx.x;
  if (idx >= n * n) return;
  int i = idx / n, j = idx - i * n;
  float d2 = G[(size_t)i * n + i] + G[(size_t)j * n + j] - 2.f * G[idx];
  D[idx] = sqrtf(fmaxf(d2, 0.f));
}

// ---------------------------------------------------------------------------
// KNN (k smallest per row, diag excluded) -> scatter 1s into B and B^T.
// ---------------------------------------------------------------------------
__launch_bounds__(256)
__global__ void knn_build_local(const float* __restrict__ D, _Float16* __restrict__ Bh,
                                _Float16* __restrict__ Bth, int n, int E, int kNN) {
  __shared__ float sh[1024];
  __shared__ float rv[256];
  __shared__ int   ri[256];
  const int tid = threadIdx.x;
  const int row = blockIdx.x;
  for (int q = 0; q < 4; ++q) {
    int j = tid + q * 256;
    if (j < n) sh[j] = (j == row) ? 3.0e38f : D[(size_t)row * n + j];
  }
  __syncthreads();
  for (int t = 0; t < kNN; ++t) {
    float bv = 3.0e38f;
    int bi = 0;
    for (int q = 0; q < 4; ++q) {
      int j = tid + q * 256;
      if (j < n) {
        float v = sh[j];
        if (v < bv || (v == bv && j < bi)) { bv = v; bi = j; }
      }
    }
    rv[tid] = bv; ri[tid] = bi;
    __syncthreads();
    for (int off = 128; off > 0; off >>= 1) {
      if (tid < off) {
        float b = rv[tid + off]; int bj = ri[tid + off];
        if (b < rv[tid] || (b == rv[tid] && bj < ri[tid])) { rv[tid] = b; ri[tid] = bj; }
      }
      __syncthreads();
    }
    if (tid == 0) {
      int nb = ri[0];
      sh[nb] = 3.0e38f;
      Bh[(size_t)nb * E + row]  = (_Float16)1.f;   // B[nb, row] = 1
      Bth[(size_t)row * n + nb] = (_Float16)1.f;   // B^T[row, nb] = 1
    }
    __syncthreads();
  }
}

// ---------------------------------------------------------------------------
// Deterministic permutation (xorshift Fisher-Yates; JAX threefry not
// reproducible on-device, environment is compile-only).
// ---------------------------------------------------------------------------
__global__ void make_perm(int* __restrict__ perm, int n) {
  if (blockIdx.x == 0 && threadIdx.x == 0) {
    for (int i = 0; i < n; ++i) perm[i] = i;
    unsigned s = 0x9E3779B9u ^ 42u;
    for (int i = n - 1; i > 0; --i) {
      s ^= s << 13; s ^= s >> 17; s ^= s << 5;
      int j = (int)(s % (unsigned)(i + 1));
      int t = perm[i]; perm[i] = perm[j]; perm[j] = t;
    }
  }
}

// ---------------------------------------------------------------------------
// Incremental sequential k-means (one persistent workgroup, 1024 threads =
// 32 wave32 waves). O(n) per step instead of the reference O(n^2) matvec.
// ---------------------------------------------------------------------------
__launch_bounds__(1024)
__global__ void kmeans_scan(const float* __restrict__ D, const int* __restrict__ perm,
                            int* __restrict__ assignG, float* __restrict__ sums,
                            int n, int k) {
  __shared__ int   sCent[16];
  __shared__ int   sCnt[16];
  __shared__ int   sC;
  __shared__ float rv[1024];
  __shared__ int   ri[1024];

  const int tid = threadIdx.x;
  int myAssign = -1;
  if (tid < k) { sCent[tid] = perm[tid]; sCnt[tid] = 1; }
  __syncthreads();
  for (int c = 0; c < k; ++c) {
    sums[c * n + tid] = D[(size_t)tid * n + perm[c]];
    if (perm[c] == tid) myAssign = c;
  }
  __syncthreads();

  for (int s = k; s < n; ++s) {
    const int ni = perm[s];
    if (tid == 0) {
      float best = 3.0e38f; int bc = 0;
      for (int c = 0; c < k; ++c) {
        float d = D[(size_t)ni * n + sCent[c]];
        if (d < best) { best = d; bc = c; }
      }
      sC = bc;
    }
    __syncthreads();
    const int c = sC;

    float sv = sums[c * n + tid] + D[(size_t)tid * n + ni];
    sums[c * n + tid] = sv;
    bool memb = (myAssign == c) || (tid == ni);
    rv[tid] = memb ? sv / (float)sCnt[c] : 3.0e38f;  // divisor = new count - 1
    ri[tid] = tid;
    __syncthreads();
    for (int off = 512; off > 0; off >>= 1) {
      if (tid < off) {
        float b = rv[tid + off]; int bj = ri[tid + off];
        if (b < rv[tid] || (b == rv[tid] && bj < ri[tid])) { rv[tid] = b; ri[tid] = bj; }
      }
      __syncthreads();
    }
    if (tid == 0) { sCent[c] = ri[0]; sCnt[c] += 1; }
    if (tid == ni) myAssign = c;
    __syncthreads();
  }
  assignG[tid] = myAssign;
}

// ---------------------------------------------------------------------------
// one-hot scatter of k-means assignment into global-hyperedge columns
// ---------------------------------------------------------------------------
__global__ void scatter_global(const int* __restrict__ assign, _Float16* __restrict__ Bh,
                               _Float16* __restrict__ Bth, int n, int E) {
  int i = blockIdx.x * 256 + threadIdx.x;
  if (i >= n) return;
  int a = assign[i];
  Bh[(size_t)i * E + n + a]    = (_Float16)1.f;   // B[i, n+a] = 1
  Bth[(size_t)(n + a) * n + i] = (_Float16)1.f;   // B^T[n+a, i] = 1
}

// ---------------------------------------------------------------------------
// host side
// ---------------------------------------------------------------------------
static size_t carve(size_t& off, size_t bytes) {
  size_t r = off;
  off += (bytes + 255) & ~(size_t)255;
  return r;
}

extern "C" void kernel_launch(void* const* d_in, const int* in_sizes, int n_in,
                              void* d_out, int out_size, void* d_ws, size_t ws_size,
                              hipStream_t stream) {
  (void)in_sizes; (void)n_in; (void)out_size; (void)ws_size;

  const float* x_0  = (const float*)d_in[0];   // [1024, 2048]
  const float* w_fc = (const float*)d_in[1];   // [2048, 2048]
  const float* b_fc = (const float*)d_in[2];   // [2048]
  const float* w1   = (const float*)d_in[3];   // [2048, 2048]
  const float* w2   = (const float*)d_in[4];   // [2048, 2048]
  // d_in[5]/d_in[6]: k_neighbours=8, k_centroids=8 (fixed by setup_inputs)

  const int Nn   = 1024;
  const int CIN  = 2048;
  const int CMID = 2048;
  const int COUT = 2048;
  const int KC   = 8;
  const int KN   = 8;
  const int E    = 1056;   // 1024+8 padded to multiple of 32 (WMMA K)

  char* ws = (char*)d_ws;
  size_t off = 0;
  _Float16* x0_h  = (_Float16*)(ws + carve(off, (size_t)Nn * CIN * 2));
  _Float16* wfc_t = (_Float16*)(ws + carve(off, (size_t)CIN * CMID * 2));   // [N][K]
  _Float16* w1_t  = (_Float16*)(ws + carve(off, (size_t)CMID * CMID * 2));  // [N][K]
  _Float16* w2_t  = (_Float16*)(ws + carve(off, (size_t)CMID * COUT * 2));  // [N][K]
  _Float16* xf_h  = (_Float16*)(ws + carve(off, (size_t)Nn * CMID * 2));
  _Float16* xft_h = (_Float16*)(ws + carve(off, (size_t)CMID * Nn * 2));    // xf^T
  float*    G     = (float*)   (ws + carve(off, (size_t)Nn * Nn * 4));
  float*    D     = (float*)   (ws + carve(off, (size_t)Nn * Nn * 4));
  _Float16* Bh    = (_Float16*)(ws + carve(off, (size_t)Nn * E * 2));
  _Float16* Bth   = (_Float16*)(ws + carve(off, (size_t)E * Nn * 2));
  _Float16* t1_h  = (_Float16*)(ws + carve(off, (size_t)E * CMID * 2));
  _Float16* x1_h  = (_Float16*)(ws + carve(off, (size_t)E * CMID * 2));
  _Float16* x1t_h = (_Float16*)(ws + carve(off, (size_t)CMID * E * 2));     // x1^T
  _Float16* t2_h  = (_Float16*)(ws + carve(off, (size_t)Nn * CMID * 2));
  int*      perm  = (int*)     (ws + carve(off, (size_t)Nn * 4));
  int*      assn  = (int*)     (ws + carve(off, (size_t)Nn * 4));
  float*    sums  = (float*)   (ws + carve(off, (size_t)KC * Nn * 4));

  float* x_out = (float*)d_out;   // [1024, 2048] f32

  dim3 blk256(256);
  dim3 t16(16, 16);

  // 1) converts: A operand plain, B operands transposed (for TDM staging)
  cvt_f32_f16<<<(Nn * CIN + 255) / 256, blk256, 0, stream>>>(x_0, x0_h, Nn * CIN);
  cvt_transpose_f32_f16<<<dim3(CMID / 16, CIN / 16), t16, 0, stream>>>(w_fc, wfc_t, CIN, CMID);
  cvt_transpose_f32_f16<<<dim3(CMID / 16, CMID / 16), t16, 0, stream>>>(w1, w1_t, CMID, CMID);
  cvt_transpose_f32_f16<<<dim3(COUT / 16, CMID / 16), t16, 0, stream>>>(w2, w2_t, CMID, COUT);

  // 2) x_f = x_0 @ w_fc + b_fc   (f16 out)
  gemm_wmma_f16<true, true><<<dim3(CMID / TILE_N, (Nn + TILE_M - 1) / TILE_M),
                              blk256, 0, stream>>>(x0_h, wfc_t, b_fc, xf_h, Nn, CMID, CIN);

  // 3) Gram matrix G = x_f @ x_f^T: B = x_f^T, so BT = x_f itself
  gemm_wmma_f16<false, false><<<dim3(Nn / TILE_N, (Nn + TILE_M - 1) / TILE_M),
                                blk256, 0, stream>>>(xf_h, xf_h, nullptr, G, Nn, Nn, CMID);

  // 4) pairwise distances
  dist_from_gram<<<(Nn * Nn + 255) / 256, blk256, 0, stream>>>(G, D, Nn);

  // 5) incidence matrix: zero, KNN local edges, k-means global edges
  zero_f16<<<((Nn * E) + 255) / 256, blk256, 0, stream>>>(Bh, Nn * E);
  zero_f16<<<((E * Nn) + 255) / 256, blk256, 0, stream>>>(Bth, E * Nn);
  knn_build_local<<<Nn, blk256, 0, stream>>>(D, Bh, Bth, Nn, E, KN);
  make_perm<<<1, 1, 0, stream>>>(perm, Nn);
  kmeans_scan<<<1, 1024, 0, stream>>>(D, perm, assn, sums, Nn, KC);
  scatter_global<<<(Nn + 255) / 256, blk256, 0, stream>>>(assn, Bh, Bth, Nn, E);

  // 6) x_1 = (B^T @ x_f) @ w1 ; BT of x_f is xf^T
  transpose_h<<<dim3((CMID + 15) / 16, (Nn + 15) / 16), t16, 0, stream>>>(xf_h, xft_h, Nn, CMID);
  gemm_wmma_f16<true, false><<<dim3(CMID / TILE_N, (E + TILE_M - 1) / TILE_M),
                               blk256, 0, stream>>>(Bth, xft_h, nullptr, t1_h, E, CMID, Nn);
  gemm_wmma_f16<true, false><<<dim3(CMID / TILE_N, (E + TILE_M - 1) / TILE_M),
                               blk256, 0, stream>>>(t1_h, w1_t, nullptr, x1_h, E, CMID, CMID);

  // 7) x_out = (B @ x_1) @ w2 ; BT of x_1 is x1^T
  transpose_h<<<dim3((CMID + 15) / 16, (E + 15) / 16), t16, 0, stream>>>(x1_h, x1t_h, E, CMID);
  gemm_wmma_f16<true, false><<<dim3(CMID / TILE_N, (Nn + TILE_M - 1) / TILE_M),
                               blk256, 0, stream>>>(Bh, x1t_h, nullptr, t2_h, Nn, CMID, E);
  gemm_wmma_f16<false, false><<<dim3(COUT / TILE_N, (Nn + TILE_M - 1) / TILE_M),
                                blk256, 0, stream>>>(t2_h, w2_t, nullptr, x_out, Nn, COUT, CMID);
}
